// GraphConvolution_38689065402671
// MI455X (gfx1250) — 16-bit and 8-bit matrix operand order repaired and checked by reading, compiled, not run on hardware
//
#include <hip/hip_runtime.h>


typedef __attribute__((ext_vector_type(8)))  int    v8i;
typedef __attribute__((ext_vector_type(8)))  float  v8f;
typedef __attribute__((ext_vector_type(16))) __bf16 v16bf;

#define N_NODES 8192
#define IN_F    1024
#define OUT_F   1024

__device__ __forceinline__ unsigned bf16_bits(float x) {
  unsigned u = __float_as_uint(x);
  return (u + 0x7FFFu + ((u >> 16) & 1u)) >> 16;
}
__device__ __forceinline__ unsigned pack_bf16x2(float lo, float hi) {
  return bf16_bits(lo) | (bf16_bits(hi) << 16);
}

union BF16Frag { v16bf v; unsigned u[8]; };

__global__ __launch_bounds__(256)
void wstats_partial(const float* __restrict__ w, float* __restrict__ part) {
  __shared__ float s1[256], s2[256];
  const int total  = IN_F * OUT_F;
  const int stride = gridDim.x * 256;
  float a = 0.f, b = 0.f;
  for (int i = blockIdx.x * 256 + threadIdx.x; i < total; i += stride) {
    float v = w[i]; a += v; b += v * v;
  }
  s1[threadIdx.x] = a; s2[threadIdx.x] = b;
  __syncthreads();
  for (int off = 128; off > 0; off >>= 1) {
    if (threadIdx.x < off) {
      s1[threadIdx.x] += s1[threadIdx.x + off];
      s2[threadIdx.x] += s2[threadIdx.x + off];
    }
    __syncthreads();
  }
  if (threadIdx.x == 0) { part[2 * blockIdx.x] = s1[0]; part[2 * blockIdx.x + 1] = s2[0]; }
}

__global__ __launch_bounds__(128)
void wstats_final(const float* __restrict__ part, float* __restrict__ stats) {
  __shared__ float s1[128], s2[128];
  s1[threadIdx.x] = part[2 * threadIdx.x];
  s2[threadIdx.x] = part[2 * threadIdx.x + 1];
  __syncthreads();
  for (int off = 64; off > 0; off >>= 1) {
    if (threadIdx.x < off) {
      s1[threadIdx.x] += s1[threadIdx.x + off];
      s2[threadIdx.x] += s2[threadIdx.x + off];
    }
    __syncthreads();
  }
  if (threadIdx.x == 0) {
    float n    = (float)(IN_F * OUT_F);
    float mean = s1[0] / n;
    float var  = (s2[0] - s1[0] * s1[0] / n) / (n - 1.f);
    stats[0] = mean;
    stats[1] = sqrtf(var);
  }
}

__global__ __launch_bounds__(256)
void quant_weight(const float* __restrict__ w, const float* __restrict__ stats,
                  const float* __restrict__ wgt_alpha, signed char* __restrict__ wqT) {
  int idx = blockIdx.x * 256 + threadIdx.x;
  float mean = stats[0], sd = stats[1];
  float wn = (w[idx] - mean) / sd;
  float wd = wn / wgt_alpha[0];
  float wc = fminf(fmaxf(wd, -1.f), 1.f);
  float q  = rintf(fabsf(wc) * 7.f);
  q = (wc < 0.f) ? -q : q;
  int k = idx >> 10;
  int n = idx & (OUT_F - 1);
  wqT[(size_t)n * IN_F + k] = (signed char)(int)q;
}

__global__ __launch_bounds__(256)
void quant_input(const float* __restrict__ x, const float* __restrict__ act_alpha,
                 unsigned char* __restrict__ xq) {
  int idx = blockIdx.x * 256 + threadIdx.x;
  float xc = fminf(x[idx] / act_alpha[0], 1.f);
  xq[idx] = (unsigned char)(int)rintf(xc * 15.f);
}

__global__ __launch_bounds__(256)
void gemm1_iu8(const unsigned char* __restrict__ xq, const signed char* __restrict__ wqT,
               const float* __restrict__ wgt_alpha, const float* __restrict__ act_alpha,
               const float* __restrict__ act_alpha2, unsigned short* __restrict__ supT) {
  const int lane = threadIdx.x & 31;
  const int wv   = threadIdx.x >> 5;
  const int hi   = lane >> 4;
  const int l16  = lane & 15;
  const int m0   = blockIdx.y * 128 + (wv >> 1) * 32;
  const int n0   = blockIdx.x * 64  + (wv & 1) * 32;

  v8i acc[2][2] = {};
  for (int k = 0; k < IN_F; k += 64) {
    v8i a[2], b[2];
#pragma unroll
    for (int i = 0; i < 2; ++i) {
      const unsigned char* pa = xq + (size_t)(m0 + 16 * i + l16) * IN_F + k + 8 * hi;
      uint2 q0 = *(const uint2*)(pa +  0);
      uint2 q1 = *(const uint2*)(pa + 16);
      uint2 q2 = *(const uint2*)(pa + 32);
      uint2 q3 = *(const uint2*)(pa + 48);
      a[i][0] = (int)q0.x; a[i][1] = (int)q0.y; a[i][2] = (int)q1.x; a[i][3] = (int)q1.y;
      a[i][4] = (int)q2.x; a[i][5] = (int)q2.y; a[i][6] = (int)q3.x; a[i][7] = (int)q3.y;
    }
#pragma unroll
    for (int j = 0; j < 2; ++j) {
      const signed char* pb = wqT + (size_t)(n0 + 16 * j + l16) * IN_F + k + 8 * hi;
      uint2 q0 = *(const uint2*)(pb +  0);
      uint2 q1 = *(const uint2*)(pb + 16);
      uint2 q2 = *(const uint2*)(pb + 32);
      uint2 q3 = *(const uint2*)(pb + 48);
      b[j][0] = (int)q0.x; b[j][1] = (int)q0.y; b[j][2] = (int)q1.x; b[j][3] = (int)q1.y;
      b[j][4] = (int)q2.x; b[j][5] = (int)q2.y; b[j][6] = (int)q3.x; b[j][7] = (int)q3.y;
    }
    acc[0][0] = __builtin_amdgcn_wmma_i32_16x16x64_iu8(false, a[0], true, b[0], acc[0][0], false, false);
    acc[0][1] = __builtin_amdgcn_wmma_i32_16x16x64_iu8(false, a[0], true, b[1], acc[0][1], false, false);
    acc[1][0] = __builtin_amdgcn_wmma_i32_16x16x64_iu8(false, a[1], true, b[0], acc[1][0], false, false);
    acc[1][1] = __builtin_amdgcn_wmma_i32_16x16x64_iu8(false, a[1], true, b[1], acc[1][1], false, false);
  }

  const float scale  = (act_alpha[0] / 15.f) * (wgt_alpha[0] / 7.f);
  const float inv_a2 = 1.f / act_alpha2[0];
#pragma unroll
  for (int i = 0; i < 2; ++i) {
#pragma unroll
    for (int j = 0; j < 2; ++j) {
      unsigned pk[4];
#pragma unroll
      for (int r = 0; r < 8; ++r) {
        float f  = (float)acc[i][j][r] * scale;
        f        = fmaxf(f, 0.f);
        float xc = fminf(f * inv_a2, 1.f);
        float sq = rintf(xc * 15.f);
        unsigned h = bf16_bits(sq);
        if (r & 1) pk[r >> 1] |= h << 16; else pk[r >> 1] = h;
      }
      const int n = n0 + 16 * j + l16;
      const int m = m0 + 16 * i + 8 * hi;
      uint4 st; st.x = pk[0]; st.y = pk[1]; st.z = pk[2]; st.w = pk[3];
      *(uint4*)(supT + (size_t)n * N_NODES + m) = st;
    }
  }
}

#define G2_M    32
#define G2_LDSW 72
__global__ __launch_bounds__(256)
void gemm2_bf16(const float* __restrict__ adj, const unsigned short* __restrict__ supT,
                const float* __restrict__ act_alpha2, float* __restrict__ out) {
  __shared__ unsigned short sa[2][G2_M * G2_LDSW];

  const int tid  = threadIdx.x;
  const int lane = tid & 31;
  const int wv   = tid >> 5;
  const int hi   = lane >> 4;
  const int l16  = lane & 15;
  const int m0   = blockIdx.x * G2_M;
  const int nw   = wv * 128;

  const int srow = tid >> 3;
  const int scol = (tid & 7) * 8;

  v8f c[2][8] = {};

  {
    const float* pa = adj + (size_t)(m0 + srow) * N_NODES + scol;
    float4 f0 = *(const float4*)(pa + 0);
    float4 f1 = *(const float4*)(pa + 4);
    unsigned* d = (unsigned*)&sa[0][srow * G2_LDSW + scol];
    d[0] = pack_bf16x2(f0.x, f0.y);
    d[1] = pack_bf16x2(f0.z, f0.w);
    d[2] = pack_bf16x2(f1.x, f1.y);
    d[3] = pack_bf16x2(f1.z, f1.w);
  }
  __syncthreads();

  const int NCH = N_NODES / 64;
  for (int kc = 0; kc < NCH; ++kc) {
    const int cur = kc & 1;
    if (kc + 1 < NCH) {
      const float* pa = adj + (size_t)(m0 + srow) * N_NODES + (kc + 1) * 64 + scol;
      float4 f0 = *(const float4*)(pa + 0);
      float4 f1 = *(const float4*)(pa + 4);
      unsigned* d = (unsigned*)&sa[cur ^ 1][srow * G2_LDSW + scol];
      d[0] = pack_bf16x2(f0.x, f0.y);
      d[1] = pack_bf16x2(f0.z, f0.w);
      d[2] = pack_bf16x2(f1.x, f1.y);
      d[3] = pack_bf16x2(f1.z, f1.w);
    }
#pragma unroll
    for (int s = 0; s < 2; ++s) {
      const int kg = kc * 64 + 32 * s;
      BF16Frag a[2];
#pragma unroll
      for (int i = 0; i < 2; ++i) {
        const unsigned short* p = &sa[cur][(16 * i + l16) * G2_LDSW + 32 * s + 8 * hi];
        uint4 q0 = *(const uint4*)(p + 0);
        uint4 q1 = *(const uint4*)(p + 16);
        a[i].u[0] = q0.x; a[i].u[1] = q0.y; a[i].u[2] = q0.z; a[i].u[3] = q0.w;
        a[i].u[4] = q1.x; a[i].u[5] = q1.y; a[i].u[6] = q1.z; a[i].u[7] = q1.w;
      }
#pragma unroll
      for (int j = 0; j < 8; ++j) {
        const unsigned short* pb = supT + (size_t)(nw + 16 * j + l16) * N_NODES + kg + 8 * hi;
        BF16Frag b;
        uint4 q0 = *(const uint4*)(pb + 0);
        uint4 q1 = *(const uint4*)(pb + 16);
        b.u[0] = q0.x; b.u[1] = q0.y; b.u[2] = q0.z; b.u[3] = q0.w;
        b.u[4] = q1.x; b.u[5] = q1.y; b.u[6] = q1.z; b.u[7] = q1.w;
        c[0][j] = __builtin_amdgcn_wmma_f32_16x16x32_bf16(false, a[0].v, false, b.v,
                                                          (short)0, c[0][j], false, false);
        c[1][j] = __builtin_amdgcn_wmma_f32_16x16x32_bf16(false, a[1].v, false, b.v,
                                                          (short)0, c[1][j], false, false);
      }
    }
    __syncthreads();
  }

  const float sd = act_alpha2[0] / 15.f;
#pragma unroll
  for (int i = 0; i < 2; ++i) {
    const int mrow = m0 + 16 * i + 8 * hi;
#pragma unroll
    for (int j = 0; j < 8; ++j) {
      const int ncol = nw + 16 * j + l16;
#pragma unroll
      for (int r = 0; r < 8; ++r)
        out[(size_t)(mrow + r) * OUT_F + ncol] = c[i][j][r] * sd;
    }
  }
}

extern "C" void kernel_launch(void* const* d_in, const int* in_sizes, int n_in,
                              void* d_out, int out_size, void* d_ws, size_t ws_size,
                              hipStream_t stream) {
  const float* input      = (const float*)d_in[0];
  const float* adj        = (const float*)d_in[1];
  const float* weight     = (const float*)d_in[3];
  const float* wgt_alpha  = (const float*)d_in[4];
  const float* act_alpha  = (const float*)d_in[5];
  const float* act_alpha2 = (const float*)d_in[6];
  float* out = (float*)d_out;

  char* ws = (char*)d_ws;
  float*          part  = (float*)(ws + 0);
  float*          stats = (float*)(ws + 4096);
  signed char*    wqT   = (signed char*)(ws + 8192);
  unsigned char*  xq    = (unsigned char*)(ws + 8192 + (size_t)IN_F * OUT_F);
  unsigned short* supT  = (unsigned short*)(ws + 8192 + (size_t)IN_F * OUT_F
                                               + (size_t)N_NODES * IN_F);

  wstats_partial<<<128, 256, 0, stream>>>(weight, part);
  wstats_final<<<1, 128, 0, stream>>>(part, stats);
  quant_weight<<<(IN_F * OUT_F) / 256, 256, 0, stream>>>(weight, stats, wgt_alpha, wqT);
  quant_input<<<(N_NODES * IN_F) / 256, 256, 0, stream>>>(input, act_alpha, xq);
  gemm1_iu8<<<dim3(OUT_F / 64, N_NODES / 128), 256, 0, stream>>>(xq, wqT, wgt_alpha, act_alpha,
                                                                 act_alpha2, supT);
  gemm2_bf16<<<N_NODES / G2_M, 256, 0, stream>>>(adj, supT, act_alpha2, out);
}
